// _ESRNN_47270410060149
// MI455X (gfx1250) — matrix operand order repaired and checked by reading, not compiled, not run on hardware
//
#include <hip/hip_runtime.h>
#include <math.h>
#include <stdint.h>

// ---------------- problem constants ----------------
#define NS      1024            // series
#define NT      1024            // time
#define SEASN   24
#define IN_SZ   24
#define OUT_SZ  24
#define EXO     6
#define HID     40
#define G4      160             // 4*HID gates
#define NW      977             // NT - IN_SZ - OUT_SZ + 1
#define KP      96              // padded K (fin + HID padded to 3*32)
#define SEASLEN (SEASN + NT)    // 1048

typedef __attribute__((ext_vector_type(16))) _Float16 v16h;
typedef __attribute__((ext_vector_type(8)))  float    v8f;

__device__ __forceinline__ float sigf(float x) { return 1.0f / (1.0f + expf(-x)); }

// ---------------- 1) exponential-smoothing scan ----------------
// one thread per series; seasonality ring realized as a flat global array
// (element t is consumed 24 steps after it was produced by the same thread,
// and same-wave store->load to the same address stays ordered).
__global__ void es_scan_kernel(const float* __restrict__ y,
                               const int*   __restrict__ idxs,
                               const float* __restrict__ lev_sms,
                               const float* __restrict__ seas_sms,
                               const float* __restrict__ init_seas,
                               float* __restrict__ seas_full,   // [NS][SEASLEN]
                               float* __restrict__ levels)      // [NS][NT] (d_out region)
{
    int s = blockIdx.x * blockDim.x + threadIdx.x;
    if (s >= NS) return;
    int sp = idxs[s];
    float lsm = sigf(lev_sms[sp]);
    float ssm = sigf(seas_sms[sp]);
    float* sf = seas_full + (size_t)s * SEASLEN;
    for (int k = 0; k < SEASN; ++k) sf[k] = expf(init_seas[sp * SEASN + k]);
    float seas0 = sf[0];
    float lev   = y[(size_t)s * NT + 0] / seas0;
    levels[(size_t)s * NT + 0] = lev;
    sf[SEASN] = seas0;                       // seasonalities[24] = seas0
    for (int t = 1; t < NT; ++t) {
        float yt = y[(size_t)s * NT + t];
        float st = sf[t];                    // ring: written 24 iters ago (or init)
        float nl = lsm * (yt / st) + (1.0f - lsm) * lev;
        float ns = ssm * (yt / nl) + (1.0f - ssm) * st;
        sf[SEASN + t] = ns;
        levels[(size_t)s * NT + t] = nl;
        lev = nl;
    }
}

// ---------------- 2) window build: windows_x (f16) + windows_y ----------------
__global__ void window_kernel(const float* __restrict__ y,
                              const float* __restrict__ cats,
                              const float* __restrict__ seas_full,
                              const float* __restrict__ levels,
                              _Float16* __restrict__ X,      // [NW][NS][30] f16
                              float* __restrict__ outY)      // [NW][NS][24]
{
    int idx = blockIdx.x * blockDim.x + threadIdx.x;
    if (idx >= NW * NS * 30) return;
    int k = idx % 30;
    int s = (idx / 30) % NS;
    int w = idx / (30 * NS);
    float lev = levels[(size_t)s * NT + w + IN_SZ];   // lev_end
    if (k < IN_SZ) {
        int tx = w + k;
        float xv = logf(y[(size_t)s * NT + tx] / seas_full[(size_t)s * SEASLEN + tx] / lev);
        X[((size_t)w * NS + s) * 30 + k] = (_Float16)xv;  // NOISE_STD=1e-3 omitted (RNG)
        int ty = w + IN_SZ + k;
        float yv = logf(y[(size_t)s * NT + ty] / seas_full[(size_t)s * SEASLEN + ty]) / lev;
        outY[((size_t)w * NS + s) * 24 + k] = yv;
    } else {
        X[((size_t)w * NS + s) * 30 + k] = (_Float16)cats[s * EXO + (k - 24)];
    }
}

// ---------------- 3) per-layer weight pack: B = [Wih || Whh] f16, N-major ----
// Bpad[n*KP + k] : k < fin -> Wih[n][k] ; k < fin+HID -> Whh[n][k-fin] ; else 0
__global__ void prep_kernel(const float* __restrict__ Wih, const float* __restrict__ Whh,
                            const float* __restrict__ bih, const float* __restrict__ bhh,
                            int fin, _Float16* __restrict__ Bpad, float* __restrict__ bias)
{
    int idx = blockIdx.x * blockDim.x + threadIdx.x;
    if (idx < G4 * KP) {
        int n = idx / KP, k = idx % KP;
        float v = 0.0f;
        if (k < fin)            v = Wih[n * fin + k];
        else if (k < fin + HID) v = Whh[n * HID + (k - fin)];
        Bpad[(size_t)n * KP + k] = (_Float16)v;
    }
    if (idx < G4) bias[idx] = bih[idx] + bhh[idx];
}

__global__ void zero_kernel(float* __restrict__ p, int n)
{
    int i = blockIdx.x * blockDim.x + threadIdx.x;
    if (i < n) p[i] = 0.0f;
}

// ---------------- 4) one dilated-LSTM time step (WMMA GEMM + cell) ----------
// Block: 256 threads (8 wave32), owns 64 batch rows. Gates [64 x 160] =
// 4 M-tiles x 10 N-tiles of 16x16, K = 96 = 3 x (16x16x32 f16 WMMA).
__global__ __launch_bounds__(256) void lstm_step_kernel(
        const _Float16* __restrict__ inx, int fin,
        const _Float16* __restrict__ Bpad,   // [G4][KP] f16 N-major
        const float*    __restrict__ bias,   // [G4]
        float* __restrict__ stH, float* __restrict__ stC,   // [Bd][HID] f32
        _Float16* __restrict__ outH,          // [NW][NS][HID] f16
        int d, int t)
{
    __shared__ __align__(16) char smem[43008];
    _Float16* ldsA = (_Float16*)smem;                 // [64][KP]  12288 B
    _Float16* ldsB = (_Float16*)(smem + 64 * KP * 2); // [G4][KP]  30720 B
    float*    ldsG = (float*)smem;                    // [64][G4]  40960 B (aliases A+B)

    const int tid   = threadIdx.x;
    const int rbase = blockIdx.x * 64;

    // ---- stage A: rows = [x (fin) | h (HID) | 0-pad], f32->f16 for h ----
    for (int i = tid; i < 64 * KP; i += 256) {
        int rl = i / KP, k = i % KP;
        int r  = rbase + rl;
        int s  = r & (NS - 1);
        int j  = r >> 10;              // r / NS
        int tm = t * d + j;
        _Float16 v = (_Float16)0.0f;
        if (k < fin) {
            if (tm < NW) v = inx[((size_t)tm * NS + s) * fin + k];  // zero-padded seq
        } else if (k < fin + HID) {
            v = (_Float16)stH[(size_t)r * HID + (k - fin)];
        }
        ldsA[i] = v;
    }
    // ---- stage B: straight dword copy of the packed weight panel ----
    {
        const uint32_t* src = (const uint32_t*)Bpad;
        uint32_t*       dst = (uint32_t*)ldsB;
        for (int i = tid; i < (G4 * KP) / 2; i += 256) dst[i] = src[i];
    }
    __syncthreads();

    const int  wv   = tid >> 5, lane = tid & 31;
    const int  mt   = wv >> 1;                // M-tile 0..3
    const int  n0   = (wv & 1) * 5;           // first of 5 N-tiles
    const bool hi   = lane >= 16;
    const int  mrow = mt * 16 + (lane & 15);

    union Frag { v16h v; uint32_t u[8]; };

    // A fragments: 16-bit A 16x32 layout (lanes0-15: K {b..b+7, b+16..b+23}, hi: +8)
    Frag a[3];
    #pragma unroll
    for (int kb = 0; kb < 3; ++kb) {
        int b0 = kb * 32 + (hi ? 8 : 0);
        const uint32_t* p0 = (const uint32_t*)(ldsA + mrow * KP + b0);
        const uint32_t* p1 = (const uint32_t*)(ldsA + mrow * KP + b0 + 16);
        #pragma unroll
        for (int q = 0; q < 4; ++q) { a[kb].u[q] = p0[q]; a[kb].u[4 + q] = p1[q]; }
    }

    v8f acc[5];
    #pragma unroll
    for (int n = 0; n < 5; ++n) {
        v8f c = {0.f, 0.f, 0.f, 0.f, 0.f, 0.f, 0.f, 0.f};
        int ncol = (n0 + n) * 16 + (lane & 15);
        #pragma unroll
        for (int kb = 0; kb < 3; ++kb) {
            Frag b;  // 16-bit B 32x16: lane = N, lanes0-15 K {0..7, 16..23}, hi: +8
            const uint32_t* pb =
                (const uint32_t*)(ldsB + (size_t)ncol * KP + kb * 32 + (hi ? 8 : 0));
            #pragma unroll
            for (int q = 0; q < 8; ++q) b.u[q] = pb[q < 4 ? q : q + 4];
            c = __builtin_amdgcn_wmma_f32_16x16x32_f16(false, a[kb].v, false, b.v,
                                                       (short)0, c, false, false);
        }
        acc[n] = c;
    }
    __syncthreads();   // all LDS A/B reads done -> safe to alias as ldsG

    // scatter accumulators (f32 C/D layout: VGPR v -> M = v + (hi?8:0))
    #pragma unroll
    for (int n = 0; n < 5; ++n) {
        int col = (n0 + n) * 16 + (lane & 15);
        #pragma unroll
        for (int v = 0; v < 8; ++v) {
            int rl = mt * 16 + v + (hi ? 8 : 0);
            ldsG[rl * G4 + col] = acc[n][v];
        }
    }
    __syncthreads();

    // ---- LSTM cell update (i,f,g,o split), write state + layer output ----
    for (int i = tid; i < 64 * HID; i += 256) {
        int rl = i / HID, hc = i % HID;
        int r  = rbase + rl;
        float gi = ldsG[rl * G4 + hc]           + bias[hc];
        float gf = ldsG[rl * G4 + HID + hc]     + bias[HID + hc];
        float gg = ldsG[rl * G4 + 2 * HID + hc] + bias[2 * HID + hc];
        float go = ldsG[rl * G4 + 3 * HID + hc] + bias[3 * HID + hc];
        float c0 = stC[(size_t)r * HID + hc];
        float cn = sigf(gf) * c0 + sigf(gi) * tanhf(gg);
        float hn = sigf(go) * tanhf(cn);
        stC[(size_t)r * HID + hc] = cn;
        stH[(size_t)r * HID + hc] = hn;
        int s  = r & (NS - 1);
        int j  = r >> 10;
        int tm = t * d + j;
        if (tm < NW) outH[((size_t)tm * NS + s) * HID + hc] = hn;
    }
}

// ---------------- 5) residual + tanh MLP + adapter head ----------------
__global__ void head_kernel(const _Float16* __restrict__ HC,
                            const _Float16* __restrict__ HB,
                            const float* __restrict__ mlp_w, const float* __restrict__ mlp_b,
                            const float* __restrict__ aw,    const float* __restrict__ ab,
                            float* __restrict__ outYhat)
{
    int idx = blockIdx.x * blockDim.x + threadIdx.x;
    if (idx >= NW * NS) return;
    const _Float16* pc = HC + (size_t)idx * HID;
    const _Float16* pb = HB + (size_t)idx * HID;
    float h[HID], z[HID];
    #pragma unroll
    for (int k = 0; k < HID; ++k) h[k] = (float)pc[k] + (float)pb[k];  // residual
    #pragma unroll
    for (int j = 0; j < HID; ++j) {
        float acc = mlp_b[j];
        #pragma unroll
        for (int k = 0; k < HID; ++k) acc += mlp_w[j * HID + k] * h[k];
        z[j] = tanhf(acc);
    }
    float* o = outYhat + (size_t)idx * OUT_SZ;
    #pragma unroll
    for (int k = 0; k < OUT_SZ; ++k) {
        float acc = ab[k];
        #pragma unroll
        for (int j = 0; j < HID; ++j) acc += aw[k * HID + j] * z[j];
        o[k] = acc;
    }
}

// ---------------- host orchestration ----------------
static inline size_t alignUp(size_t x, size_t a) { return (x + a - 1) & ~(a - 1); }

extern "C" void kernel_launch(void* const* d_in, const int* in_sizes, int n_in,
                              void* d_out, int out_size, void* d_ws, size_t ws_size,
                              hipStream_t stream)
{
    (void)in_sizes; (void)n_in; (void)out_size; (void)ws_size;
    // inputs in setup_inputs() dict order (params flattened in insertion order)
    const float* y         = (const float*)d_in[0];
    const int*   idxs      = (const int*)  d_in[1];
    const float* cats      = (const float*)d_in[2];
    const float* lev_sms   = (const float*)d_in[3];
    const float* seas_sms  = (const float*)d_in[4];
    const float* init_seas = (const float*)d_in[5];
    const float* mlp_w     = (const float*)d_in[6];
    const float* mlp_b     = (const float*)d_in[7];
    const float* aw        = (const float*)d_in[8];
    const float* ab        = (const float*)d_in[9];
    // lstm_{gi}_{di}_{Wih,Whh,bih,bhh} starting at index 10, 4 per layer

    // workspace carve-out (256-B aligned)
    char* ws = (char*)d_ws;
    size_t off = 0;
    auto take = [&](size_t bytes) -> char* {
        char* p = ws + off; off = alignUp(off + bytes, 256); return p;
    };
    float*    seas = (float*)   take((size_t)NS * SEASLEN * 4);
    _Float16* X    = (_Float16*)take((size_t)NW * NS * 30 * 2);
    _Float16* HA   = (_Float16*)take((size_t)NW * NS * HID * 2);
    _Float16* HB   = (_Float16*)take((size_t)NW * NS * HID * 2);
    _Float16* HC   = (_Float16*)take((size_t)NW * NS * HID * 2);
    float*    stH  = (float*)   take((size_t)24 * NS * HID * 4);  // max Bd = 24*1024
    float*    stC  = (float*)   take((size_t)24 * NS * HID * 4);
    _Float16* Bp[4]; float* bias[4];
    for (int L = 0; L < 4; ++L) {
        Bp[L]   = (_Float16*)take((size_t)G4 * KP * 2);
        bias[L] = (float*)   take((size_t)G4 * 4);
    }

    float* outY    = (float*)d_out;                       // [NW][NS][24]
    float* outYhat = outY    + (size_t)NW * NS * 24;      // [NW][NS][24]
    float* outLev  = outYhat + (size_t)NW * NS * 24;      // [NS][NT]

    // 1) ES scan (writes levels output + seasonality buffer)
    es_scan_kernel<<<dim3((NS + 255) / 256), dim3(256), 0, stream>>>(
        y, idxs, lev_sms, seas_sms, init_seas, seas, outLev);

    // 2) window build
    {
        int total = NW * NS * 30;
        window_kernel<<<dim3((total + 255) / 256), dim3(256), 0, stream>>>(
            y, cats, seas, outLev, X, outY);
    }

    // 3) pack weights per layer
    const int fins[4] = {30, 40, 40, 40};
    for (int L = 0; L < 4; ++L) {
        const float* Wih = (const float*)d_in[10 + 4 * L + 0];
        const float* Whh = (const float*)d_in[10 + 4 * L + 1];
        const float* bih = (const float*)d_in[10 + 4 * L + 2];
        const float* bhh = (const float*)d_in[10 + 4 * L + 3];
        prep_kernel<<<dim3((G4 * KP + 255) / 256), dim3(256), 0, stream>>>(
            Wih, Whh, bih, bhh, fins[L], Bp[L], bias[L]);
    }

    // 4) dilated LSTM stack: L0(d=1) X->HA, L1(d=4) HA->HB,
    //    L2(d=8) HB->HA, L3(d=24) HA->HC ; residual = HB (group-0 output)
    struct Layer { int fin, d, steps; const _Float16* in; _Float16* out; };
    Layer Ls[4] = {
        {30, 1,  (NW + 0)  / 1,        X,  HA},   // 977 steps
        {40, 4,  (NW + 3)  / 4,        HA, HB},   // 245
        {40, 8,  (NW + 7)  / 8,        HB, HA},   // 123
        {40, 24, (NW + 23) / 24,       HA, HC},   // 41
    };
    for (int L = 0; L < 4; ++L) {
        int Bd = Ls[L].d * NS;
        int nst = Bd * HID;
        zero_kernel<<<dim3((nst + 255) / 256), dim3(256), 0, stream>>>(stH, nst);
        zero_kernel<<<dim3((nst + 255) / 256), dim3(256), 0, stream>>>(stC, nst);
        for (int t = 0; t < Ls[L].steps; ++t) {
            lstm_step_kernel<<<dim3(Bd / 64), dim3(256), 0, stream>>>(
                Ls[L].in, Ls[L].fin, Bp[L], bias[L], stH, stC, Ls[L].out,
                Ls[L].d, t);
        }
    }

    // 5) residual + MLP + adapter head
    {
        int total = NW * NS;
        head_kernel<<<dim3((total + 255) / 256), dim3(256), 0, stream>>>(
            HC, HB, mlp_w, mlp_b, aw, ab, outYhat);
    }
}
